// MultiheadAttention_57930518889148
// MI455X (gfx1250) — matrix operand order repaired and checked by reading, not compiled, not run on hardware
//
#include <hip/hip_runtime.h>
#include <hip/hip_bf16.h>

// ---- shapes ----
#define SEQ   2048
#define DM    1024
#define NH    16
#define DK    64
#define BB    2
#define MTOT  (BB * SEQ)   // 4096 rows for the big GEMMs

typedef __attribute__((ext_vector_type(16))) __bf16          v16bf;
typedef __attribute__((ext_vector_type(8)))  float           v8f;
typedef __attribute__((ext_vector_type(8)))  unsigned short  ushort8;
typedef __attribute__((ext_vector_type(4)))  float           v4f;

static __device__ __forceinline__ unsigned short f2bf(float f) {
  unsigned int u = __float_as_uint(f);
  u += 0x7FFFu + ((u >> 16) & 1u);          // round-to-nearest-even
  return (unsigned short)(u >> 16);
}
static __device__ __forceinline__ float bf2f(unsigned short h) {
  return __uint_as_float(((unsigned int)h) << 16);
}

union FragU { v16bf v; ushort8 u8[2]; unsigned short s[16]; };

// A-matrix 16x32 bf16 fragment (ISA 7.12.2): lane m(0-15): K = [k0..k0+7],[k0+16..k0+23]
//                                            lane m+16:    K = [k0+8..k0+15],[k0+24..k0+31]
static __device__ __forceinline__ v16bf load_a_bf16(const unsigned short* row, int lane) {
  const unsigned short* p = row + ((lane & 16) ? 8 : 0);
  FragU f;
  f.u8[0] = *(const ushort8*)(p);
  f.u8[1] = *(const ushort8*)(p + 16);
  return f.v;
}
// B-matrix 32x16 bf16 fragment: lane n(0-15): K = [k0..k0+7],[k0+16..k0+23]; lane n+16: K = [k0+8..k0+15],[k0+24..k0+31]
static __device__ __forceinline__ v16bf load_b_bf16(const unsigned short* row, int lane) {
  const unsigned short* p = row + ((lane & 16) ? 8 : 0);
  FragU f;
  f.u8[0] = *(const ushort8*)(p);
  f.u8[1] = *(const ushort8*)(p + 16);
  return f.v;
}
// Same layouts but source data is f32 (convert to bf16 in-register)
static __device__ __forceinline__ v16bf load_a_f32(const float* row, int lane) {
  const float* p = row + ((lane & 16) ? 8 : 0);
  v4f x0 = *(const v4f*)(p);      v4f x1 = *(const v4f*)(p + 4);
  v4f x2 = *(const v4f*)(p + 16); v4f x3 = *(const v4f*)(p + 20);
  FragU f;
#pragma unroll
  for (int i = 0; i < 4; ++i) {
    f.s[i]      = f2bf(x0[i]);  f.s[4 + i]  = f2bf(x1[i]);
    f.s[8 + i]  = f2bf(x2[i]);  f.s[12 + i] = f2bf(x3[i]);
  }
  return f.v;
}
static __device__ __forceinline__ v16bf load_b_f32(const float* row, int lane) {
  const float* p = row + ((lane & 16) ? 8 : 0);
  v4f x0 = *(const v4f*)(p);     v4f x1 = *(const v4f*)(p + 4);
  v4f x2 = *(const v4f*)(p + 16); v4f x3 = *(const v4f*)(p + 20);
  FragU f;
#pragma unroll
  for (int i = 0; i < 4; ++i) {
    f.s[i]      = f2bf(x0[i]);  f.s[4 + i]  = f2bf(x1[i]);
    f.s[8 + i]  = f2bf(x2[i]);  f.s[12 + i] = f2bf(x3[i]);
  }
  return f.v;
}

static __device__ __forceinline__ v8f wmma_bf16(v16bf a, v16bf b, v8f c) {
  // D = A x B + C, 16x16x32 bf16 -> f32
  return __builtin_amdgcn_wmma_f32_16x16x32_bf16(false, a, false, b, (short)0, c,
                                                 false, false);
}

__device__ const v8f kZero8 = {0.f, 0.f, 0.f, 0.f, 0.f, 0.f, 0.f, 0.f};

// ---------------------------------------------------------------------------
// Projection GEMM: Out[m,n] = bf16( sum_k X[m,k]*W[n,k] + bias[n] )
// M=4096, N=1024, K=1024. VT=0: row-major Out. VT=1: per-head transposed
// store for V: Vt[b,h,d,s] so A*V's B-matrix rows are contiguous.
// ---------------------------------------------------------------------------
template <int VT>
__global__ __launch_bounds__(128) void proj_kernel(const float* __restrict__ X,
                                                   const float* __restrict__ W,
                                                   const float* __restrict__ bias,
                                                   unsigned short* __restrict__ Out) {
  const int lane = threadIdx.x & 31;
  const int wave = threadIdx.x >> 5;
  const int m0 = blockIdx.x * 64 + (wave & 1) * 32;
  const int n0 = blockIdx.y * 64 + (wave >> 1) * 32;

  const float* arow0 = X + (size_t)(m0 + (lane & 15)) * DM;
  const float* arow1 = arow0 + (size_t)16 * DM;
  const float* brow0 = W + (size_t)(n0 + (lane & 15)) * DM;
  const float* brow1 = brow0 + (size_t)16 * DM;

  v8f acc[2][2];
#pragma unroll
  for (int i = 0; i < 2; ++i)
#pragma unroll
    for (int j = 0; j < 2; ++j) acc[i][j] = kZero8;

  for (int k0 = 0; k0 < DM; k0 += 32) {
    __builtin_prefetch(arow0 + k0 + 64, 0, 0);
    __builtin_prefetch(brow0 + k0 + 64, 0, 0);
    v16bf a0 = load_a_f32(arow0 + k0, lane);
    v16bf a1 = load_a_f32(arow1 + k0, lane);
    v16bf b0 = load_b_f32(brow0 + k0, lane);
    v16bf b1 = load_b_f32(brow1 + k0, lane);
    acc[0][0] = wmma_bf16(a0, b0, acc[0][0]);
    acc[0][1] = wmma_bf16(a0, b1, acc[0][1]);
    acc[1][0] = wmma_bf16(a1, b0, acc[1][0]);
    acc[1][1] = wmma_bf16(a1, b1, acc[1][1]);
  }

  const int cn = lane & 15;
  const int mo = (lane >> 4) << 3;   // lanes 0-15 -> M rows r, lanes 16-31 -> r+8
#pragma unroll
  for (int ti = 0; ti < 2; ++ti) {
#pragma unroll
    for (int tj = 0; tj < 2; ++tj) {
      const int gnc = n0 + tj * 16 + cn;
      const float bval = bias[gnc];
#pragma unroll
      for (int r = 0; r < 8; ++r) {
        const int gm = m0 + ti * 16 + mo + r;
        const float val = acc[ti][tj][r] + bval;
        if (VT == 0) {
          Out[(size_t)gm * DM + gnc] = f2bf(val);
        } else {
          const int b = gm >> 11, s = gm & (SEQ - 1);
          const int h = gnc >> 6, d = gnc & (DK - 1);
          Out[(((size_t)(b * NH + h) * DK + d) << 11) + s] = f2bf(val);
        }
      }
    }
  }
}

// one 32-key score block: 4 WMMA with hoisted Q fragments
static __device__ __forceinline__ void score_block(v16bf a_d0, v16bf a_d1,
                                                   const unsigned short* krow0,
                                                   const unsigned short* krow1,
                                                   int lane, v8f& acc0, v8f& acc1) {
  acc0 = kZero8;
  acc1 = kZero8;
  acc0 = wmma_bf16(a_d0, load_b_bf16(krow0, lane), acc0);
  acc0 = wmma_bf16(a_d1, load_b_bf16(krow0 + 32, lane), acc0);
  acc1 = wmma_bf16(a_d0, load_b_bf16(krow1, lane), acc1);
  acc1 = wmma_bf16(a_d1, load_b_bf16(krow1 + 32, lane), acc1);
}

// ---------------------------------------------------------------------------
// Fused attention core: per (b,h, 16-query strip) one wave keeps the full
// causal score strip (16 x 2048 bf16 = 64 KB) in LDS.
//   P1: scores = QK^T/8 (WMMA) -> LDS; mask-free loop below the diagonal,
//       branchless bit-select mask only for the one diagonal block
//   P2: row max, e = exp(s-max) back into LDS, row sum       (all intra-wave)
//   P3: H = (e . V) * (1/sum)  with A-fragments served from LDS (WMMA)
//   P4: attn = e * (1/sum) streamed once with non-temporal b128 stores
// attn HBM traffic: exactly one 512 MB (NT) write instead of 2 GB of passes.
// ---------------------------------------------------------------------------
__global__ __launch_bounds__(32) void flash_kernel(const unsigned short* __restrict__ Qb,
                                                   const unsigned short* __restrict__ Kb,
                                                   const unsigned short* __restrict__ Vt,
                                                   float* __restrict__ attn,
                                                   unsigned short* __restrict__ Hb) {
  extern __shared__ unsigned short sm[];          // 16 rows x SEQ bf16 = 64 KB
  const int z = blockIdx.z;                       // b*NH + h
  const int b = z >> 4, h = z & (NH - 1);
  const int q0 = blockIdx.x * 16;
  const int lane = threadIdx.x & 31;
  const int r16 = lane & 15;
  const int cn = lane & 15;
  const int mo = (lane >> 4) << 3;

  const int kbCount = (q0 + 16 + 31) >> 5;        // 32-key blocks covering keys <= q0+15
  const int kbFull  = (q0 + 1) >> 5;              // blocks entirely below the diagonal
  const int kend = kbCount << 5;

  // ---- Phase 1: scores into LDS --------------------------------------------
  const unsigned short* qrow = Qb + (size_t)(b * SEQ + q0 + r16) * DM + h * DK;
  const v16bf a_d0 = load_a_bf16(qrow, lane);     // Q fragments reused for all keys
  const v16bf a_d1 = load_a_bf16(qrow + 32, lane);
  const unsigned short* kbase = Kb + (size_t)b * SEQ * DM + h * DK;

  int kb = 0;
  for (; kb < kbFull; ++kb) {                     // mask-free region
    const int kcol0 = kb << 5;
    const unsigned short* krow0 = kbase + (size_t)(kcol0 + r16) * DM;
    v8f acc0, acc1;
    score_block(a_d0, a_d1, krow0, krow0 + (size_t)16 * DM, lane, acc0, acc1);
#pragma unroll
    for (int tj = 0; tj < 2; ++tj) {
      const v8f acc = tj ? acc1 : acc0;
      const int col = kcol0 + tj * 16 + cn;
#pragma unroll
      for (int r = 0; r < 8; ++r) sm[(mo + r) * SEQ + col] = f2bf(acc[r] * 0.125f);
    }
  }
  for (; kb < kbCount; ++kb) {                    // diagonal blocks: branchless mask
    const int kcol0 = kb << 5;
    const unsigned short* krow0 = kbase + (size_t)(kcol0 + r16) * DM;
    v8f acc0, acc1;
    score_block(a_d0, a_d1, krow0, krow0 + (size_t)16 * DM, lane, acc0, acc1);
#pragma unroll
    for (int tj = 0; tj < 2; ++tj) {
      const v8f acc = tj ? acc1 : acc0;
      const int col = kcol0 + tj * 16 + cn;
#pragma unroll
      for (int r = 0; r < 8; ++r) {
        const int row = mo + r;
        const unsigned int sv = (unsigned int)f2bf(acc[r] * 0.125f);
        const unsigned int keep = 0u - (unsigned int)(col <= q0 + row); // all-ones if valid
        sm[row * SEQ + col] = (unsigned short)((sv & keep) | (0xFF80u & ~keep));
      }
    }
  }

  // ---- Phase 2: row max, exp, row sum (lane pairs split each row) ----------
  unsigned short* lrow = sm + r16 * SEQ;
  float m = -3.402823466e38f;
  for (int c = (lane >> 4) * 8; c < kend; c += 16) {
    ushort8 s8 = *(const ushort8*)(lrow + c);
#pragma unroll
    for (int i = 0; i < 8; ++i) m = fmaxf(m, bf2f(s8[i]));
  }
  m = fmaxf(m, __shfl_xor(m, 16, 32));

  float sum = 0.f;
  for (int c = (lane >> 4) * 8; c < kend; c += 16) {
    ushort8 s8 = *(const ushort8*)(lrow + c);
    ushort8 e8;
#pragma unroll
    for (int i = 0; i < 8; ++i) {
      const float e = __expf(bf2f(s8[i]) - m);    // exp(-inf - m) == 0 => mask
      sum += e;
      e8[i] = f2bf(e);
    }
    *(ushort8*)(lrow + c) = e8;
  }
  sum += __shfl_xor(sum, 16, 32);
  const float inv = 1.0f / sum;

  // ---- Phase 3: H = (e . V) * inv, A-fragments from LDS --------------------
  v8f o[4];
#pragma unroll
  for (int tj = 0; tj < 4; ++tj) o[tj] = kZero8;
  const unsigned short* vbase = Vt + (size_t)z * DK * SEQ;
  const unsigned short* vrow0 = vbase + (size_t)(0 * 16 + r16) * SEQ;
  const unsigned short* vrow1 = vbase + (size_t)(1 * 16 + r16) * SEQ;
  const unsigned short* vrow2 = vbase + (size_t)(2 * 16 + r16) * SEQ;
  const unsigned short* vrow3 = vbase + (size_t)(3 * 16 + r16) * SEQ;
  const unsigned short* arow = sm + r16 * SEQ;

  for (int k0 = 0; k0 < kend; k0 += 32) {
    v16bf a = load_a_bf16(arow + k0, lane);
    o[0] = wmma_bf16(a, load_b_bf16(vrow0 + k0, lane), o[0]);
    o[1] = wmma_bf16(a, load_b_bf16(vrow1 + k0, lane), o[1]);
    o[2] = wmma_bf16(a, load_b_bf16(vrow2 + k0, lane), o[2]);
    o[3] = wmma_bf16(a, load_b_bf16(vrow3 + k0, lane), o[3]);
  }

  float invr[8];
#pragma unroll
  for (int r = 0; r < 8; ++r) invr[r] = __shfl(inv, mo + r, 32);
#pragma unroll
  for (int tj = 0; tj < 4; ++tj) {
#pragma unroll
    for (int r = 0; r < 8; ++r) {
      const int gm = q0 + mo + r;
      Hb[(size_t)(b * SEQ + gm) * DM + h * DK + tj * 16 + cn] = f2bf(o[tj][r] * invr[r]);
    }
  }

  // ---- Phase 4: stream normalized probabilities (non-temporal, write-once) -
  {
    const int half = lane >> 4;
    const int base = half * 1024;
    float* gout = attn + (size_t)z * SEQ * SEQ + (size_t)(q0 + r16) * SEQ + base;
    const unsigned short* erow = lrow + base;
    int nvalid = kend - base;
    nvalid = nvalid < 0 ? 0 : (nvalid > 1024 ? 1024 : nvalid);   // multiple of 8
    int c = 0;
    for (; c < nvalid; c += 8) {
      ushort8 e8 = *(const ushort8*)(erow + c);
      v4f o0, o1;
#pragma unroll
      for (int i = 0; i < 4; ++i) {
        o0[i] = bf2f(e8[i]) * inv;
        o1[i] = bf2f(e8[4 + i]) * inv;
      }
      __builtin_nontemporal_store(o0, (v4f*)(gout + c));
      __builtin_nontemporal_store(o1, (v4f*)(gout + c + 4));
    }
    const v4f z4 = {0.f, 0.f, 0.f, 0.f};
    for (; c < 1024; c += 8) {                    // fully-masked tail: zeros
      __builtin_nontemporal_store(z4, (v4f*)(gout + c));
      __builtin_nontemporal_store(z4, (v4f*)(gout + c + 4));
    }
  }
}

// ---------------------------------------------------------------------------
// Output projection: out[m,n] = sum_k H[m,k]*Wo[n,k] + bo[n]  (f32 result)
// ---------------------------------------------------------------------------
__global__ __launch_bounds__(128) void outproj_kernel(const unsigned short* __restrict__ Hb,
                                                      const float* __restrict__ Wo,
                                                      const float* __restrict__ bo,
                                                      float* __restrict__ out) {
  const int lane = threadIdx.x & 31;
  const int wave = threadIdx.x >> 5;
  const int m0 = blockIdx.x * 64 + (wave & 1) * 32;
  const int n0 = blockIdx.y * 64 + (wave >> 1) * 32;

  const unsigned short* arow0 = Hb + (size_t)(m0 + (lane & 15)) * DM;
  const unsigned short* arow1 = arow0 + (size_t)16 * DM;
  const float* brow0 = Wo + (size_t)(n0 + (lane & 15)) * DM;
  const float* brow1 = brow0 + (size_t)16 * DM;

  v8f acc[2][2];
#pragma unroll
  for (int i = 0; i < 2; ++i)
#pragma unroll
    for (int j = 0; j < 2; ++j) acc[i][j] = kZero8;

  for (int k0 = 0; k0 < DM; k0 += 32) {
    __builtin_prefetch(brow0 + k0 + 64, 0, 0);
    v16bf a0 = load_a_bf16(arow0 + k0, lane);
    v16bf a1 = load_a_bf16(arow1 + k0, lane);
    v16bf b0 = load_b_f32(brow0 + k0, lane);
    v16bf b1 = load_b_f32(brow1 + k0, lane);
    acc[0][0] = wmma_bf16(a0, b0, acc[0][0]);
    acc[0][1] = wmma_bf16(a0, b1, acc[0][1]);
    acc[1][0] = wmma_bf16(a1, b0, acc[1][0]);
    acc[1][1] = wmma_bf16(a1, b1, acc[1][1]);
  }

  const int cn = lane & 15;
  const int mo = (lane >> 4) << 3;
#pragma unroll
  for (int ti = 0; ti < 2; ++ti) {
#pragma unroll
    for (int tj = 0; tj < 2; ++tj) {
      const int gn = n0 + tj * 16 + cn;
      const float bval = bo[gn];
#pragma unroll
      for (int r = 0; r < 8; ++r) {
        const int gm = m0 + ti * 16 + mo + r;
        __builtin_nontemporal_store(acc[ti][tj][r] + bval,
                                    out + (size_t)gm * DM + gn);
      }
    }
  }
}

// ---------------------------------------------------------------------------
extern "C" void kernel_launch(void* const* d_in, const int* in_sizes, int n_in,
                              void* d_out, int out_size, void* d_ws, size_t ws_size,
                              hipStream_t stream) {
  (void)in_sizes; (void)n_in; (void)out_size; (void)ws_size;
  const float* q   = (const float*)d_in[0];
  const float* k   = (const float*)d_in[1];
  const float* v   = (const float*)d_in[2];
  // d_in[3] = causal mask (bool) -- handled analytically (k > q => masked)
  const float* w_q = (const float*)d_in[4];
  const float* b_q = (const float*)d_in[5];
  const float* w_k = (const float*)d_in[6];
  const float* b_k = (const float*)d_in[7];
  const float* w_v = (const float*)d_in[8];
  const float* b_v = (const float*)d_in[9];
  const float* w_o = (const float*)d_in[10];
  const float* b_o = (const float*)d_in[11];

  float* out  = (float*)d_out;                       // (B,S,D)   : 4M f32
  float* attn = out + (size_t)BB * SEQ * DM;         // (B,H,S,S) : 134M f32

  // workspace: Qb, Kb (B*S, D) bf16; Vt (B,H,DK,S) bf16; Hb (B*S, D) bf16 = 32 MB
  unsigned short* Qb = (unsigned short*)d_ws;
  unsigned short* Kb = Qb + (size_t)MTOT * DM;
  unsigned short* Vt = Kb + (size_t)MTOT * DM;
  unsigned short* Hb = Vt + (size_t)BB * NH * DK * SEQ;

  dim3 blk(128);
  dim3 gproj(MTOT / 64, DM / 64);
  proj_kernel<0><<<gproj, blk, 0, stream>>>(q, w_q, b_q, Qb);
  proj_kernel<0><<<gproj, blk, 0, stream>>>(k, w_k, b_k, Kb);
  proj_kernel<1><<<gproj, blk, 0, stream>>>(v, w_v, b_v, Vt);

  // one wave per 16-query strip; 64 KB dynamic LDS holds the score strip
  dim3 gfa(SEQ / 16, 1, BB * NH);
  flash_kernel<<<gfa, dim3(32), 16 * SEQ * sizeof(unsigned short), stream>>>(
      Qb, Kb, Vt, attn, Hb);

  outproj_kernel<<<gproj, blk, 0, stream>>>(Hb, w_o, b_o, out);
}
